// BatchChildSumTreeLSTM_67302137528486
// MI455X (gfx1250) — matrix operand order repaired and checked by reading, not compiled, not run on hardware
//
#include <hip/hip_runtime.h>
#include <math.h>
#include <stdint.h>

// ---------------------------------------------------------------------------
// Child-Sum TreeLSTM on gfx1250 (MI455X): level-by-level, WMMA f16 GEMMs with
// fp32 accumulate. Child-h tiles staged LDS via the Tensor Data Mover (TDM).
// ---------------------------------------------------------------------------

typedef __attribute__((ext_vector_type(8)))  _Float16 v8h;
typedef __attribute__((ext_vector_type(16))) _Float16 v16h;
typedef __attribute__((ext_vector_type(8)))  float    v8f;
typedef __attribute__((ext_vector_type(4)))  unsigned int u32x4;
typedef __attribute__((ext_vector_type(8)))  int      i32x8;
typedef __attribute__((ext_vector_type(4)))  int      i32x4;

#define HID   128
#define PITCH 136          // LDS row pitch in halfs (128 + 8 pad; 16B aligned, bank-friendly)
#define WROW  16384        // halfs per 128x128 weight matrix

__device__ __forceinline__ v16h cat8(v8h a, v8h b) {
  return __builtin_shufflevector(a, b, 0,1,2,3,4,5,6,7,8,9,10,11,12,13,14,15);
}
__device__ __forceinline__ float sigmoidf_(float x) { return 1.0f / (1.0f + __expf(-x)); }

__device__ __forceinline__ v8f wmma16x16x32(v16h a, v16h b, v8f c) {
  // D = A(16x32 f16) * B(32x16 f16) + C(16x16 f32)
  return __builtin_amdgcn_wmma_f32_16x16x32_f16(false, a, false, b, (short)0, c, false, false);
}

// TDM: DMA a [rows x 128] f16 tile (row stride 128) from global into LDS at
// byte offset lds_off, padded +16B per 256B row => LDS pitch 136 halfs.
// Rows beyond `rows` are OOB and return zero per the D# spec.
__device__ __forceinline__ void tdm_load_child_tile(unsigned lds_off,
                                                    const _Float16* gsrc, int rows) {
  unsigned long long ga = (unsigned long long)(uintptr_t)(const void*)gsrc;
  // D# group0: count=1 | lds_addr | global_addr(57b) | type=2
  u32x4 g0 = { 1u,
               lds_off,
               (unsigned)(ga & 0xFFFFFFFFu),
               (unsigned)((ga >> 32) & 0x01FFFFFFu) | 0x80000000u };
  // D# group1: wg_mask=0, data_size=1(2B), pad_enable=1, pad_interval=5(64 dw),
  // pad_amount=3(4 dw); tensor_dim0=128, tensor_dim1=rows, tile 128x256,
  // tensor_dim0_stride=128.
  i32x8 g1 = { (int)((1u << 16) | (1u << 20) | (5u << 22) | (3u << 25)),
               (int)(128u << 16),                 // tensor_dim0[15:0] @ bits63:48
               (int)((unsigned)(rows & 0xFFFF) << 16), // tensor_dim1[15:0] @ bits95:80
               (int)((rows >> 16) | (128 << 16)), // tensor_dim1 hi | tile_dim0=128
               256,                               // tile_dim1=256 (tile_dim2=0)
               128,                               // tensor_dim0_stride low32
               0, 0 };
  i32x4 gz = { 0, 0, 0, 0 };
#if defined(__clang_major__) && (__clang_major__ >= 23)
  i32x8 gz8 = { 0, 0, 0, 0, 0, 0, 0, 0 };
  __builtin_amdgcn_tensor_load_to_lds(g0, g1, gz, gz, gz8, 0);
#else
  __builtin_amdgcn_tensor_load_to_lds(g0, g1, gz, gz, 0);
#endif
}

// Convert the 8 gate weight matrices fp32 -> f16, row-major (layout already
// matches the contiguous per-lane B-fragment runs, so no repacking needed).
__global__ void convert_weights_kernel(const float* __restrict__ Wix, const float* __restrict__ Wih,
                                       const float* __restrict__ Wox, const float* __restrict__ Woh,
                                       const float* __restrict__ Wux, const float* __restrict__ Wuh,
                                       const float* __restrict__ Wfx, const float* __restrict__ Wfh,
                                       _Float16* __restrict__ dst) {
  int i = blockIdx.x * 256 + threadIdx.x;            // 8 * 16384 total
  const float* srcs[8] = {Wix, Wih, Wox, Woh, Wux, Wuh, Wfx, Wfh};
  int m = i >> 14, j = i & (WROW - 1);
  dst[i] = (_Float16)srcs[m][j];
}

// One TreeLSTM level. Block = 128 threads = 4 wave32; each wave owns 16 nodes.
// HC = this level has children (non-leaf).
template <bool HC>
__global__ __launch_bounds__(128)
void tree_level_kernel(const float* __restrict__ x, int n,
                       const _Float16* __restrict__ ch_h, const float* __restrict__ ch_c,
                       _Float16* __restrict__ out_h, float* __restrict__ out_c,
                       const _Float16* __restrict__ w16,
                       const float* __restrict__ bix, const float* __restrict__ bxo,
                       const float* __restrict__ bxu, const float* __restrict__ bxf) {
  extern __shared__ char smem[];
  _Float16* xs   = (_Float16*)smem;                  // 64 x PITCH halfs (x tile, f16)
  _Float16* hsum = xs + 64 * PITCH;                  // 64 x PITCH halfs (sum of child h), HC only
  _Float16* chh  = hsum + 64 * PITCH;                // 256 x PITCH halfs (child h tile, TDM), HC only
  float* fxbuf   = (float*)(chh + 256 * PITCH);      // 4 waves x 16x16 f32 (fx preact), HC only
  float* fcbuf   = fxbuf + 4 * 256;                  // 4 waves x 16x16 f32 (fc partial), HC only

  const int t = threadIdx.x;
  const int block0 = blockIdx.x * 64;

  // ---- async: TDM the 256x128 f16 child-h tile into LDS (wave 0 issues) ----
  if (HC) {
    if (t < 32) {
      int rows = n * 4 - block0 * 4;
      if (rows > 256) rows = 256;
      tdm_load_child_tile((unsigned)(uintptr_t)(void*)chh,
                          ch_h + (size_t)block0 * 4 * HID, rows);
    }
  }

  // ---- x tile: global f32 -> LDS f16 (overlaps with the TDM transfer) -----
  for (int idx = t; idx < 64 * HID; idx += 128) {
    int r = idx >> 7, k = idx & 127;
    int node = block0 + r;
    float v = (node < n) ? x[(size_t)node * HID + k] : 0.0f;
    xs[r * PITCH + k] = (_Float16)v;
  }

  if (HC) {
    if (t < 32) __builtin_amdgcn_s_wait_tensorcnt(0);  // only issuing wave tracks TENSORcnt
    __syncthreads();
    // h_sum from the LDS child tile (rows beyond n*4 were OOB-zeroed by TDM)
    for (int idx = t; idx < 64 * HID; idx += 128) {
      int r = idx >> 7, k = idx & 127;
      const _Float16* c0 = chh + (4 * r) * PITCH + k;
      float s = (float)c0[0] + (float)c0[PITCH] + (float)c0[2 * PITCH] + (float)c0[3 * PITCH];
      hsum[r * PITCH + k] = (_Float16)s;
    }
  }
  __syncthreads();

  const int wave  = t >> 5;
  const int lane  = t & 31;
  const int tile0 = block0 + wave * 16;
  if (tile0 >= n) return;                            // wave-uniform: EXEC stays full for WMMA

  const int col  = lane & 15;                        // N / hidden column within tile
  const int hi   = (lane >= 16);
  const int aoff = hi ? 16 : 0;                      // A-fragment K sub-offset (halfs)
  const int boff = hi ? 16 : 0;                      // B-fragment K sub-offset (halfs)
  const int row8 = hi ? 8 : 0;                       // C/D row offset

  // A fragments for x and h_sum (4 K-blocks of 32): two 16B LDS reads each.
  v16h ax[4], ah[4];
#pragma unroll
  for (int kb = 0; kb < 4; ++kb) {
    const _Float16* p = xs + (wave * 16 + col) * PITCH + kb * 32 + aoff;
    ax[kb] = cat8(*(const v8h*)p, *(const v8h*)(p + 8));
    if (HC) {
      const _Float16* q = hsum + (wave * 16 + col) * PITCH + kb * 32 + aoff;
      ah[kb] = cat8(*(const v8h*)q, *(const v8h*)(q + 8));
    }
  }

  const _Float16* Wix = w16 + 0 * WROW;
  const _Float16* Wih = w16 + 1 * WROW;
  const _Float16* Wox = w16 + 2 * WROW;
  const _Float16* Woh = w16 + 3 * WROW;
  const _Float16* Wux = w16 + 4 * WROW;
  const _Float16* Wuh = w16 + 5 * WROW;
  const _Float16* Wfx = w16 + 6 * WROW;
  const _Float16* Wfh = w16 + 7 * WROW;

  float* fxw = fxbuf + wave * 256;
  float* fcw = fcbuf + wave * 256;

  for (int nt = 0; nt < 8; ++nt) {                   // 8 N-tiles of 16 hidden units
    v8f ai = {}, ao = {}, au = {}, af = {};
    const int wr = (nt * 16 + col) * HID;            // weight row base (halfs); B[k][n] = W[n][k]
#pragma unroll
    for (int kb = 0; kb < 4; ++kb) {
      const int wo = wr + kb * 32 + boff;
      // Batch all gate B-fragment loads for this K-block so the loads clause
      // together and overlap the WMMA issue instead of wait-lockstepping.
      v16h b_ix = cat8(*(const v8h*)(Wix + wo), *(const v8h*)(Wix + wo + 8));
      v16h b_ox = cat8(*(const v8h*)(Wox + wo), *(const v8h*)(Wox + wo + 8));
      v16h b_ux = cat8(*(const v8h*)(Wux + wo), *(const v8h*)(Wux + wo + 8));
      if (HC) {
        v16h b_ih = cat8(*(const v8h*)(Wih + wo), *(const v8h*)(Wih + wo + 8));
        v16h b_oh = cat8(*(const v8h*)(Woh + wo), *(const v8h*)(Woh + wo + 8));
        v16h b_uh = cat8(*(const v8h*)(Wuh + wo), *(const v8h*)(Wuh + wo + 8));
        v16h b_fx = cat8(*(const v8h*)(Wfx + wo), *(const v8h*)(Wfx + wo + 8));
        ai = wmma16x16x32(ax[kb], b_ix, ai);
        ao = wmma16x16x32(ax[kb], b_ox, ao);
        au = wmma16x16x32(ax[kb], b_ux, au);
        af = wmma16x16x32(ax[kb], b_fx, af);
        ai = wmma16x16x32(ah[kb], b_ih, ai);
        ao = wmma16x16x32(ah[kb], b_oh, ao);
        au = wmma16x16x32(ah[kb], b_uh, au);
      } else {
        ai = wmma16x16x32(ax[kb], b_ix, ai);
        ao = wmma16x16x32(ax[kb], b_ox, ao);
        au = wmma16x16x32(ax[kb], b_ux, au);
      }
    }

    const float bi = bix[nt * 16 + col];
    const float bo = bxo[nt * 16 + col];
    const float bu = bxu[nt * 16 + col];

    float fc[8];
#pragma unroll
    for (int r = 0; r < 8; ++r) fc[r] = 0.0f;

    if (HC) {
      // fx preact to per-wave LDS (needed indexed by child row), zero fc buffer.
      const float bf = bxf[nt * 16 + col];
#pragma unroll
      for (int r = 0; r < 8; ++r) {
        fxw[(r + row8) * 16 + col] = af[r] + bf;
        fcw[(r + row8) * 16 + col] = 0.0f;
      }
      asm volatile("s_wait_dscnt 0" ::: "memory");   // same-wave LDS RAW fence

      // fh = child_h @ Wfh^T for the 64 child rows (4 tiles of 16), fused with
      // f = sigmoid(fx + bfx + fh); fc_parent += f * child_c.
#pragma unroll
      for (int ct = 0; ct < 4; ++ct) {
        v8f afh = {};
        const int crloc = wave * 64 + ct * 16 + col; // block-local child row
#pragma unroll
        for (int kb = 0; kb < 4; ++kb) {
          const _Float16* pa = chh + crloc * PITCH + kb * 32 + aoff;
          v16h a = cat8(*(const v8h*)pa, *(const v8h*)(pa + 8));
          const int wo = wr + kb * 32 + boff;
          v16h b = cat8(*(const v8h*)(Wfh + wo), *(const v8h*)(Wfh + wo + 8));
          afh = wmma16x16x32(a, b, afh);
        }
        const int p0 = 4 * ct + (hi ? 2 : 0);        // first parent this lane covers
        float acc0 = 0.0f, acc1 = 0.0f;
#pragma unroll
        for (int r = 0; r < 8; ++r) {
          const int   mloc = ct * 16 + r + row8;     // child row within wave tile
          const int   pl   = mloc >> 2;              // its parent (local)
          const float fpre = fxw[pl * 16 + col] + afh[r];
          const float fv   = sigmoidf_(fpre);
          const float cc   = ch_c[((size_t)tile0 * 4 + mloc) * HID + nt * 16 + col];
          if (r < 4) acc0 += fv * cc; else acc1 += fv * cc;
        }
        fcw[p0 * 16 + col]       += acc0;            // unique (parent,col) per lane
        fcw[(p0 + 1) * 16 + col] += acc1;
      }
      asm volatile("s_wait_dscnt 0" ::: "memory");
#pragma unroll
      for (int r = 0; r < 8; ++r) fc[r] = fcw[(r + row8) * 16 + col];
    }

    // Finish gates, write h (f16, consumed by parent GEMMs) and c (f32).
#pragma unroll
    for (int r = 0; r < 8; ++r) {
      const int   node = tile0 + r + row8;
      const float iv = sigmoidf_(ai[r] + bi);
      const float ov = sigmoidf_(ao[r] + bo);
      const float uv = tanhf(au[r] + bu);
      const float cv = iv * uv + fc[r];
      const float hv = ov * tanhf(cv);
      if (node < n) {
        const size_t oidx = (size_t)node * HID + nt * 16 + col;
        out_c[oidx] = cv;
        out_h[oidx] = (_Float16)hv;
      }
    }
  }
}

// Final projection: out[16,5] = h_root @ Wout^T + bout  (tiny -> plain VALU)
__global__ void tree_out_kernel(const _Float16* __restrict__ h, const float* __restrict__ Wout,
                                const float* __restrict__ bout, float* __restrict__ out) {
  int i = threadIdx.x;
  if (i >= 80) return;
  int node = i / 5, l = i % 5;
  float s = bout[l];
  for (int k = 0; k < HID; ++k) s += (float)h[node * HID + k] * Wout[l * HID + k];
  out[i] = s;
}

extern "C" void kernel_launch(void* const* d_in, const int* in_sizes, int n_in,
                              void* d_out, int out_size, void* d_ws, size_t ws_size,
                              hipStream_t stream) {
  const float* embeds = (const float*)d_in[0];
  const float* Wix  = (const float*)d_in[1];
  const float* bix  = (const float*)d_in[2];
  const float* Wih  = (const float*)d_in[3];
  const float* Wfx  = (const float*)d_in[4];
  const float* bfx  = (const float*)d_in[5];
  const float* Wfh  = (const float*)d_in[6];
  const float* Wox  = (const float*)d_in[7];
  const float* box_ = (const float*)d_in[8];
  const float* Woh  = (const float*)d_in[9];
  const float* Wux  = (const float*)d_in[10];
  const float* bux  = (const float*)d_in[11];
  const float* Wuh  = (const float*)d_in[12];
  const float* Wout = (const float*)d_in[13];
  const float* bout = (const float*)d_in[14];

  static const int  counts[8] = {16, 64, 256, 1024, 4096, 16384, 65536, 262144};
  static const long offs[9]   = {0, 16, 80, 336, 1360, 5456, 21840, 87376, 349520};

  // Workspace layout: [f16 weights 256KB][h f16, all nodes ~85MB][c f32, all nodes ~171MB]
  char* ws = (char*)d_ws;
  _Float16* w16     = (_Float16*)ws;
  _Float16* h16_all = (_Float16*)(ws + 262144);
  float*    c_all   = (float*)(ws + 262144 + (size_t)349520 * 128 * 2);

  // HC kernel needs ~110KB dynamic LDS (chh tile staged by TDM): raise the cap.
  const size_t smemHC = (size_t)(2 * 64 * PITCH + 256 * PITCH) * sizeof(_Float16)
                      + (size_t)(2 * 4 * 256) * sizeof(float);
  (void)hipFuncSetAttribute((const void*)&tree_level_kernel<true>,
                            hipFuncAttributeMaxDynamicSharedMemorySize, (int)smemHC);

  convert_weights_kernel<<<512, 256, 0, stream>>>(Wix, Wih, Wox, Woh, Wux, Wuh, Wfx, Wfh, w16);

  for (int level = 7; level >= 0; --level) {
    const int n = counts[level];
    const float* xlvl = embeds + offs[level] * 128;
    _Float16* oh = h16_all + offs[level] * 128;
    float*    oc = c_all + offs[level] * 128;
    dim3 grid((n + 63) / 64), block(128);
    if (level == 7) {
      size_t smem = (size_t)64 * PITCH * sizeof(_Float16);
      tree_level_kernel<false><<<grid, block, smem, stream>>>(
          xlvl, n, nullptr, nullptr, oh, oc, w16, bix, box_, bux, bfx);
    } else {
      const _Float16* ch = h16_all + offs[level + 1] * 128;
      const float*    cc = c_all + offs[level + 1] * 128;
      tree_level_kernel<true><<<grid, block, smemHC, stream>>>(
          xlvl, n, ch, cc, oh, oc, w16, bix, box_, bux, bfx);
    }
  }

  tree_out_kernel<<<1, 128, 0, stream>>>(h16_all /* roots at node offset 0 */, Wout, bout, (float*)d_out);
}
